// KernelGenerator_72370198938102
// MI455X (gfx1250) — matrix operand order repaired and checked by reading, not compiled, not run on hardware
//
#include <hip/hip_runtime.h>
#include <hip/hip_bf16.h>
#include <math.h>

// ---------------------------------------------------------------------------
// MI455X (gfx1250, wave32) implementation.
//   - Attention 1x1 (K=128) and 3x3 (K=576) convs run as implicit GEMMs on
//     v_wmma_f32_16x16x32_bf16 (bf16 in, f32 accumulate).
//   - Weights are prepacked to bf16 once, then each GEMM block pulls its
//     weight tile into LDS with a single TDM tensor_load_to_lds 2D descriptor
//     (rows x 256 bf16), synchronized via s_wait_tensorcnt + barrier.
//   - All tensors in workspace use channel-major [ch][pixel] layout where
//     pixel P = b*16384 + h*128 + w, so every VALU kernel is fully coalesced.
//   - ~3.5 GFLOPs vs ~100MB traffic: memory-bound at 23.3 TB/s, so the GEMMs
//     on the matrix pipes are effectively free and the rest is streaming.
// ---------------------------------------------------------------------------

typedef __attribute__((ext_vector_type(16))) __bf16 v16bf;
typedef __attribute__((ext_vector_type(8)))  float  v8f;

#define NPIX 32768   // b*h*w
#define HWSZ 16384   // h*w
#define CCH  64

__device__ __forceinline__ float leakyf(float v) { return v >= 0.f ? v : 0.2f * v; }

// ------------------------- TDM weight staging ------------------------------
#if defined(__has_builtin)
#if __has_builtin(__builtin_amdgcn_tensor_load_to_lds)
#define HAVE_TDM 1
#endif
#endif

#ifdef HAVE_TDM
typedef __attribute__((ext_vector_type(4))) unsigned int u32x4;
typedef __attribute__((ext_vector_type(8))) int          i32x8;
typedef __attribute__((ext_vector_type(4))) int          i32x4;

// 2D TDM copy: rows x 256 bf16 elements, contiguous in memory and LDS.
__device__ __forceinline__ void tdm_load_rows_to_lds(const __bf16* gsrc,
                                                     void* lds, unsigned rows) {
  unsigned long long ga = (unsigned long long)(const void*)gsrc;
  unsigned int la = (unsigned int)(unsigned long long)lds;   // addr[31:0] = LDS offset
  u32x4 g0;
  g0.x = 1u;                                                  // count=1 (valid user D#)
  g0.y = la;                                                  // lds_addr
  g0.z = (unsigned int)ga;                                    // global_addr[31:0]
  g0.w = (unsigned int)((ga >> 32) & 0x1FFFFFFull) | (2u << 30); // addr[56:32] | type=2
  i32x8 g1;
  g1[0] = (int)(1u << 16);        // workgroup_mask=0, data_size=1 (2 bytes)
  g1[1] = (int)(256u << 16);      // tensor_dim0 = 256 (low 16 -> dw1[31:16])
  g1[2] = (int)(rows << 16);      // tensor_dim0 hi = 0 | tensor_dim1 low16 = rows
  g1[3] = (int)(256u << 16);      // tensor_dim1 hi = 0 | tile_dim0 = 256
  g1[4] = (int)rows;              // tile_dim1 = rows, tile_dim2 = 0
  g1[5] = 256;                    // tensor_dim0_stride = 256
  g1[6] = 0;                      // stride hi | tensor_dim1_stride lo
  g1[7] = 0;
  i32x4 gz = {0, 0, 0, 0};
#if defined(__clang_major__) && (__clang_major__ >= 23)
  i32x8 gz8 = {0, 0, 0, 0, 0, 0, 0, 0};
  __builtin_amdgcn_tensor_load_to_lds(g0, g1, gz, gz, gz8, 0);
#else
  __builtin_amdgcn_tensor_load_to_lds(g0, g1, gz, gz, 0);
#endif
}
#endif

// Stage `rows*256` bf16 weights into LDS (TDM fast path, copy fallback).
__device__ __forceinline__ void stage_weights(const __bf16* __restrict__ gsrc,
                                              __bf16* lds, int rows, int tid) {
#ifdef HAVE_TDM
  if (tid < 32) {                       // one wave issues the DMA
    tdm_load_rows_to_lds(gsrc, lds, (unsigned)rows);
    __builtin_amdgcn_s_wait_tensorcnt(0);
  }
  __syncthreads();
#else
  const int n = rows * 256;
  for (int i = tid; i < n; i += 256) lds[i] = gsrc[i];
  __syncthreads();
#endif
}

// ------------------------- weight prepack (f32 -> bf16) --------------------
// p1: [oc 0..63][k 0..127]            (attn conv1, layout already [n][k])
// p2: [oc 0..63][tap*64+ic]           (attn conv2, tap-major K)
__global__ __launch_bounds__(256) void k_prepack(
    const float* __restrict__ w1, const float* __restrict__ w2,
    __bf16* __restrict__ p1, __bf16* __restrict__ p2)
{
  const int i = blockIdx.x * 256 + threadIdx.x;
  if (i < 64 * 128) p1[i] = (__bf16)w1[i];
  if (i < 64 * 576) {
    const int oc = i / 576, r = i % 576;
    const int tap = r / 64, ic = r % 64;
    p2[oc * 576 + tap * 64 + ic] = (__bf16)w2[oc * 576 + ic * 9 + tap];
  }
}

// ------------------------- attn conv1: 1x1, 128 -> 64, relu ----------------
// GEMM: D[m=pixel][n=outc] = sum_k A[m][k] * W1[n][k],  K = 128 (x then y).
__global__ __launch_bounds__(256) void k_attn_conv1(
    const float* __restrict__ x, const float* __restrict__ y,
    const __bf16* __restrict__ p1, const float* __restrict__ b1,
    float* __restrict__ t1)
{
  __shared__ __align__(32) __bf16 Wl[64 * 128];          // [n][k] bf16, 16KB
  const int tid = threadIdx.x;
  stage_weights(p1, Wl, 32, tid);                        // 32 rows x 256 bf16

  const int lane  = tid & 31;
  const int wave  = tid >> 5;
  const int P0    = blockIdx.x * 128 + wave * 16;        // 16-pixel tile
  const int row   = lane & 15;                           // A row / B,D column
  const int khalf = lane >> 4;
  const int P     = P0 + row;
  const int b     = P >> 14;
  const int hw    = P & (HWSZ - 1);
  const float* xb = x + (size_t)b * (CCH * HWSZ) + hw;
  const float* yb = y + (size_t)b * (CCH * HWSZ) + hw;

  v8f acc[4] = {};
  for (int ks = 0; ks < 4; ++ks) {
    v16bf a;
#pragma unroll
    for (int e = 0; e < 16; ++e) {
      const int k = ks * 32 + khalf * 16 + e;
      const float v = (k < 64) ? xb[k * HWSZ] : yb[(k - 64) * HWSZ];
      a[e] = (__bf16)v;
    }
#pragma unroll
    for (int nt = 0; nt < 4; ++nt) {
      const v16bf bf = *(const v16bf*)&Wl[(nt * 16 + row) * 128 + ks * 32 + khalf * 16];
      acc[nt] = __builtin_amdgcn_wmma_f32_16x16x32_bf16(
          false, a, false, bf, (short)0, acc[nt], false, false);
    }
  }
#pragma unroll
  for (int nt = 0; nt < 4; ++nt) {
    const int oc = nt * 16 + row;
    const float bias = b1[oc];
#pragma unroll
    for (int r = 0; r < 8; ++r) {
      const int Pm = P0 + khalf * 8 + r;
      float v = acc[nt][r] + bias;
      t1[oc * NPIX + Pm] = fmaxf(v, 0.f);
    }
  }
}

// ------------------------- attn conv2: 3x3, 64 -> 64, relu -----------------
// Implicit GEMM, K = 9*64 = 576, k = tap*64 + ic (tap fixed per 32-wide step).
__global__ __launch_bounds__(256) void k_attn_conv2(
    const float* __restrict__ t1, const __bf16* __restrict__ p2,
    const float* __restrict__ b2, float* __restrict__ t2)
{
  __shared__ __align__(32) __bf16 Wl[32 * 576];          // [oc][tap*64+ic], 36KB
  const int tid    = threadIdx.x;
  const int ocbase = blockIdx.y * 32;
  stage_weights(p2 + (size_t)ocbase * 576, Wl, 72, tid); // 72 rows x 256 bf16

  const int lane  = tid & 31;
  const int wave  = tid >> 5;
  const int P0    = blockIdx.x * 128 + wave * 16;
  const int row   = lane & 15;
  const int khalf = lane >> 4;
  const int P     = P0 + row;
  const int b     = P >> 14;
  const int hw    = P & (HWSZ - 1);
  const int y0    = hw >> 7;
  const int x0    = hw & 127;
  const float* tb = t1 + b * HWSZ;

  v8f acc[2] = {};
  for (int ks = 0; ks < 18; ++ks) {
    const int tap = ks >> 1;
    const int ky = tap / 3, kx = tap % 3;
    const int py = y0 + ky - 1, px = x0 + kx - 1;
    const bool inb = (py >= 0) & (py < 128) & (px >= 0) & (px < 128);
    const int off = py * 128 + px;
    v16bf a;
#pragma unroll
    for (int e = 0; e < 16; ++e) {
      const int ic = (ks & 1) * 32 + khalf * 16 + e; // 0..63
      const float v = inb ? tb[ic * NPIX + off] : 0.f;
      a[e] = (__bf16)v;
    }
#pragma unroll
    for (int nt = 0; nt < 2; ++nt) {
      const v16bf bf = *(const v16bf*)&Wl[(nt * 16 + row) * 576 + ks * 32 + khalf * 16];
      acc[nt] = __builtin_amdgcn_wmma_f32_16x16x32_bf16(
          false, a, false, bf, (short)0, acc[nt], false, false);
    }
  }
#pragma unroll
  for (int nt = 0; nt < 2; ++nt) {
    const int oc = ocbase + nt * 16 + row;
    const float bias = b2[oc];
#pragma unroll
    for (int r = 0; r < 8; ++r) {
      const int Pm = P0 + khalf * 8 + r;
      float v = acc[nt][r] + bias;
      t2[oc * NPIX + Pm] = fmaxf(v, 0.f);
    }
  }
}

// ------------------------- attn conv3: 1x1 -> 2, sigmoid -------------------
__global__ __launch_bounds__(256) void k_attn3(
    const float* __restrict__ t2, const float* __restrict__ w3,
    const float* __restrict__ b3, float* __restrict__ attn)
{
  const int P = blockIdx.x * 256 + threadIdx.x;
  float a0 = b3[0], a1 = b3[1];
  for (int ic = 0; ic < 64; ++ic) {
    const float v = t2[ic * NPIX + P];
    a0 += w3[ic] * v;
    a1 += w3[64 + ic] * v;
  }
  attn[P]        = 1.f / (1.f + __expf(-a0));
  attn[NPIX + P] = 1.f / (1.f + __expf(-a1));
}

// ------------------------- gate x,y; relayout to [c][P] --------------------
__global__ __launch_bounds__(256) void k_gate(
    const float* __restrict__ x, const float* __restrict__ y,
    const float* __restrict__ attn, float* __restrict__ xg, float* __restrict__ yg)
{
  const int id = blockIdx.x * 256 + threadIdx.x;     // c*NPIX + P
  const int c = id >> 15, P = id & (NPIX - 1);
  const int b = P >> 14, hw = P & (HWSZ - 1);
  const size_t src = (size_t)b * (CCH * HWSZ) + (size_t)c * HWSZ + hw;
  xg[id] = x[src] * attn[P];
  yg[id] = y[src] * attn[NPIX + P];
}

// ------------------------- depthwise kxk conv ------------------------------
__global__ __launch_bounds__(256) void k_dw(
    const float* __restrict__ in, float* __restrict__ out,
    const float* __restrict__ w, const float* __restrict__ bias,
    int C, int k, int pad)
{
  const int id = blockIdx.x * 256 + threadIdx.x;
  const int c = id >> 15, P = id & (NPIX - 1);
  if (c >= C) return;
  const int b = P >> 14, hw = P & (HWSZ - 1);
  const int y0 = hw >> 7, x0 = hw & 127;
  const float* ib = in + c * NPIX + b * HWSZ;
  const float* wc = w + c * k * k;
  float acc = bias[c];
  for (int ky = 0; ky < k; ++ky) {
    const int py = y0 + ky - pad;
    if (py < 0 || py >= 128) continue;
    for (int kx = 0; kx < k; ++kx) {
      const int px = x0 + kx - pad;
      if (px < 0 || px >= 128) continue;
      acc += wc[ky * k + kx] * ib[py * 128 + px];
    }
  }
  out[c * NPIX + P] = acc;
}

// ------------------------- pointwise 1x1 conv ------------------------------
__global__ __launch_bounds__(256) void k_pw(
    const float* __restrict__ in, float* __restrict__ out,
    const float* __restrict__ w, const float* __restrict__ bias,
    int Cin, int Cout, int do_leaky, int outOff)
{
  const int id = blockIdx.x * 256 + threadIdx.x;
  const int co = id >> 15, P = id & (NPIX - 1);
  if (co >= Cout) return;
  float acc = bias[co];
  const float* wr = w + co * Cin;
  for (int ci = 0; ci < Cin; ++ci) acc += wr[ci] * in[ci * NPIX + P];
  if (do_leaky) acc = leakyf(acc);
  out[(co + outOff) * NPIX + P] = acc;
}

// ------------------------- global average pool -----------------------------
__global__ __launch_bounds__(256) void k_gap(const float* __restrict__ xg,
                                             float* __restrict__ g)
{
  __shared__ float red[256];
  const int b = blockIdx.x & 1, c = blockIdx.x >> 1;
  const float* p = xg + c * NPIX + b * HWSZ;
  float s = 0.f;
  for (int j = threadIdx.x; j < HWSZ; j += 256) s += p[j];
  red[threadIdx.x] = s;
  __syncthreads();
  for (int st = 128; st; st >>= 1) {
    if (threadIdx.x < st) red[threadIdx.x] += red[threadIdx.x + st];
    __syncthreads();
  }
  if (threadIdx.x == 0) g[b * 64 + c] = red[0] * (1.f / 16384.f);
}

// ------------------------- spectral MLP stage A ----------------------------
__global__ void k_spec_a(const float* __restrict__ g, const float* __restrict__ wa,
                         const float* __restrict__ ba, float* __restrict__ ga)
{
  const int t = threadIdx.x;          // 64 threads: b = t>>5, m = t&31
  const int b = t >> 5, m = t & 31;
  float acc = ba[m];
  for (int c = 0; c < 64; ++c) acc += wa[m * 64 + c] * g[b * 64 + c];
  ga[b * 32 + m] = leakyf(acc);
}

// ------------------------- spectral MLP stage B ----------------------------
__global__ __launch_bounds__(128) void k_spec_b(
    const float* __restrict__ wb, const float* __restrict__ bb,
    const float* __restrict__ ga, float* __restrict__ se, int k2, int off)
{
  const int id = blockIdx.x * 128 + threadIdx.x;
  const int total = 2 * 64 * k2;
  if (id >= total) return;
  const int b = id / (64 * k2);
  const int r = id % (64 * k2);
  const int c = r / k2, j = r % k2;
  const int o = c * k2 + j;
  float acc = bb[o];
  for (int m = 0; m < 32; ++m) acc += wb[o * 32 + m] * ga[b * 32 + m];
  se[(b * 64 + c) * 35 + off + j] = acc;
}

// ------------------------- KernelNorm over 35 spatial channels -------------
__global__ __launch_bounds__(256) void k_norm_sp(float* __restrict__ sp)
{
  const int P = blockIdx.x * 256 + threadIdx.x;
  float v[35];
  float mean = 0.f;
#pragma unroll
  for (int j = 0; j < 35; ++j) { v[j] = sp[j * NPIX + P]; mean += v[j]; }
  mean *= (1.f / 35.f);
  float var = 0.f;
#pragma unroll
  for (int j = 0; j < 35; ++j) { const float d = v[j] - mean; var += d * d; }
  var *= (1.f / 34.f);                       // ddof = 1
  const float inv = 1.f / (sqrtf(var) + 1e-10f);
#pragma unroll
  for (int j = 0; j < 35; ++j) sp[j * NPIX + P] = (v[j] - mean) * inv;
}

// ------------------------- KernelNorm over 35 spectral dims ----------------
__global__ void k_norm_se(float* __restrict__ se)
{
  const int id = threadIdx.x;                // 128 threads: (b,c)
  float* p = se + id * 35;
  float mean = 0.f;
  for (int j = 0; j < 35; ++j) mean += p[j];
  mean *= (1.f / 35.f);
  float var = 0.f;
  for (int j = 0; j < 35; ++j) { const float d = p[j] - mean; var += d * d; }
  var *= (1.f / 34.f);
  const float inv = 1.f / (sqrtf(var) + 1e-10f);
  for (int j = 0; j < 35; ++j) p[j] = (p[j] - mean) * inv;
}

// ------------------------- dynamic-kernel apply ----------------------------
__global__ __launch_bounds__(256) void k_apply(
    const float* __restrict__ xg, const float* __restrict__ sp,
    const float* __restrict__ se, float* __restrict__ out,
    int k, int pad, int off)
{
  __shared__ float sk[25];
  const int b = blockIdx.z, c = blockIdx.y;
  const int tid = threadIdx.x;
  const int k2 = k * k;
  if (tid < k2) sk[tid] = se[(b * 64 + c) * 35 + off + tid];
  __syncthreads();
  const int hw = blockIdx.x * 256 + tid;
  const int y0 = hw >> 7, x0 = hw & 127;
  const int Pp = b * HWSZ + hw;
  const float* xc = xg + c * NPIX + b * HWSZ;
  __builtin_prefetch(xc + hw, 0, 3);         // global_prefetch_b8
  float acc = 0.f;
  for (int ky = 0; ky < k; ++ky) {
    const int py = y0 + ky - pad;
    for (int kx = 0; kx < k; ++kx) {
      const int px = x0 + kx - pad;
      const int j = ky * k + kx;
      const float xv = (py >= 0 && py < 128 && px >= 0 && px < 128)
                           ? xc[py * 128 + px] : 0.f;
      acc += xv * sk[j] * sp[(off + j) * NPIX + Pp];
    }
  }
  out[(size_t)b * (CCH * HWSZ) + (size_t)c * HWSZ + hw] = acc;
}

// ---------------------------------------------------------------------------
extern "C" void kernel_launch(void* const* d_in, const int* in_sizes, int n_in,
                              void* d_out, int out_size, void* d_ws, size_t ws_size,
                              hipStream_t stream) {
  (void)in_sizes; (void)n_in; (void)out_size;
  const float* x   = (const float*)d_in[0];
  const float* y   = (const float*)d_in[1];
  const float* aw1 = (const float*)d_in[2];
  const float* ab1 = (const float*)d_in[3];
  const float* aw2 = (const float*)d_in[4];
  const float* ab2 = (const float*)d_in[5];
  const float* aw3 = (const float*)d_in[6];
  const float* ab3 = (const float*)d_in[7];

  // Workspace layout (floats)
  float* ws = (float*)d_ws;
  const size_t T1 = 0;                // [64][32768]
  const size_t T2 = T1 + 2097152;
  const size_t XG = T2 + 2097152;
  const size_t YG = XG + 2097152;
  const size_t AT = YG + 2097152;     // [2][32768]
  const size_t SA = AT + 65536;       // [<=64][32768] scratch ping
  const size_t SB = SA + 2097152;     // pong
  const size_t SP = SB + 2097152;     // [35][32768]
  const size_t G_ = SP + 1146880;     // [2][64]
  const size_t GA = G_ + 128;         // [2][32]
  const size_t SE = GA + 64;          // [2][64][35]
  const size_t WP = SE + 4480;        // bf16 prepacked weights (45056 elems)
  const size_t NEED = WP + 22784;
  if (ws_size < NEED * sizeof(float)) return;

  float* t1 = ws + T1; float* t2 = ws + T2;
  float* xg = ws + XG; float* yg = ws + YG;
  float* at = ws + AT; float* sa = ws + SA; float* sb = ws + SB;
  float* sp = ws + SP; float* g  = ws + G_; float* ga = ws + GA;
  float* se = ws + SE;
  __bf16* p1 = (__bf16*)(ws + WP);     // [64][128]
  __bf16* p2 = p1 + 64 * 128;          // [64][576] tap-major

  // --- prepack weights to bf16, then attention path (WMMA GEMMs) ---
  k_prepack<<<144, 256, 0, stream>>>(aw1, aw2, p1, p2);
  k_attn_conv1<<<256, 256, 0, stream>>>(x, y, p1, ab1, t1);
  k_attn_conv2<<<dim3(256, 2), 256, 0, stream>>>(t1, p2, ab2, t2);
  k_attn3<<<128, 256, 0, stream>>>(t2, aw3, ab3, at);
  k_gate<<<8192, 256, 0, stream>>>(x, y, at, xg, yg);
  k_gap<<<128, 256, 0, stream>>>(xg, g);

  const int KS_[3]  = {1, 3, 5};
  const int PAD_[3] = {0, 1, 2};
  const int OFF_[3] = {0, 1, 10};

  for (int i = 0; i < 3; ++i) {
    const int k = KS_[i], pad = PAD_[i], k2 = k * k, off = OFF_[i];
    const int sbi = 8 + 12 * i;                 // spatial params base index
    const float* dw1 = (const float*)d_in[sbi + 0];
    const float* db1 = (const float*)d_in[sbi + 1];
    const float* pw1 = (const float*)d_in[sbi + 2];
    const float* pb1 = (const float*)d_in[sbi + 3];
    const float* dw2 = (const float*)d_in[sbi + 4];
    const float* db2 = (const float*)d_in[sbi + 5];
    const float* pw2 = (const float*)d_in[sbi + 6];
    const float* pb2 = (const float*)d_in[sbi + 7];
    const float* dw3 = (const float*)d_in[sbi + 8];
    const float* db3 = (const float*)d_in[sbi + 9];
    const float* pw3 = (const float*)d_in[sbi + 10];
    const float* pb3 = (const float*)d_in[sbi + 11];
    const int qbi = 44 + 4 * i;                 // spectral params base index
    const float* wa = (const float*)d_in[qbi + 0];
    const float* ba = (const float*)d_in[qbi + 1];
    const float* wb = (const float*)d_in[qbi + 2];
    const float* bb = (const float*)d_in[qbi + 3];

    // spatial chain on gated y
    k_dw<<<64 * 128, 256, 0, stream>>>(yg, sa, dw1, db1, 64, k, pad);
    k_pw<<<k2 * 128, 256, 0, stream>>>(sa, sb, pw1, pb1, 64, k2, 0, 0);
    k_dw<<<k2 * 128, 256, 0, stream>>>(sb, sa, dw2, db2, k2, k, pad);
    k_pw<<<k2 * 128, 256, 0, stream>>>(sa, sb, pw2, pb2, k2, k2, 1, 0);
    k_dw<<<k2 * 128, 256, 0, stream>>>(sb, sa, dw3, db3, k2, k, pad);
    k_pw<<<k2 * 128, 256, 0, stream>>>(sa, sp, pw3, pb3, k2, k2, 0, off);

    // spectral MLP
    k_spec_a<<<1, 64, 0, stream>>>(g, wa, ba, ga);
    const int tot = 2 * 64 * k2;
    k_spec_b<<<(tot + 127) / 128, 128, 0, stream>>>(wb, bb, ga, se, k2, off);
  }

  // KernelNorms
  k_norm_sp<<<128, 256, 0, stream>>>(sp);
  k_norm_se<<<1, 128, 0, stream>>>(se);

  // dynamic-kernel apply -> three outputs concatenated
  float* out = (float*)d_out;
  for (int i = 0; i < 3; ++i) {
    const int k = KS_[i], pad = PAD_[i], off = OFF_[i];
    k_apply<<<dim3(64, 64, 2), 256, 0, stream>>>(
        xg, sp, se, out + (size_t)i * 2097152, k, pad, off);
  }
}
